// GAT_14980845928643
// MI455X (gfx1250) — matrix operand order repaired and checked by reading, not compiled, not run on hardware
//
#include <hip/hip_runtime.h>
#include <hip/hip_bf16.h>

typedef __attribute__((ext_vector_type(16))) _Float16 v16h;
typedef __attribute__((ext_vector_type(8)))  _Float16 v8h;
typedef __attribute__((ext_vector_type(8)))  float    v8f;
typedef __attribute__((ext_vector_type(4)))  int      v4i;

#define NEG_SLOPE 0.2f

#if __has_builtin(__builtin_amdgcn_global_load_async_to_lds_b128) && \
    __has_builtin(__builtin_amdgcn_s_wait_asynccnt)
#define HAVE_ASYNC_LDS 1
#endif

// ---------- helpers ----------
__device__ __forceinline__ unsigned fenc(float f) {
    unsigned u = __float_as_uint(f);
    return (u & 0x80000000u) ? ~u : (u | 0x80000000u);
}
__device__ __forceinline__ float fdec(unsigned u) {
    unsigned v = (u & 0x80000000u) ? (u & 0x7FFFFFFFu) : ~u;
    return __uint_as_float(v);
}
#define ENC_NEG_INF 0x007FFFFFu  // fenc(-inf)

// async 16B global->LDS copy (CDNA5 GLOBAL_LOAD_ASYNC_TO_LDS_B128), sync fallback
__device__ __forceinline__ void copy16_g2l(void* lds_dst, const void* gsrc) {
#ifdef HAVE_ASYNC_LDS
    __builtin_amdgcn_global_load_async_to_lds_b128(
        (__attribute__((address_space(1))) v4i*)(uintptr_t)gsrc,
        (__attribute__((address_space(3))) v4i*)(uint32_t)(uintptr_t)lds_dst, 0, 0);
#else
    *(uint4*)lds_dst = *(const uint4*)gsrc;
#endif
}
__device__ __forceinline__ void copy_wait() {
#ifdef HAVE_ASYNC_LDS
    __builtin_amdgcn_s_wait_asynccnt(0);
#endif
}

// ---------- tiny utility kernels ----------
__global__ void k_fill_f32(float* __restrict__ p, float v, int n) {
    int i = blockIdx.x * blockDim.x + threadIdx.x;
    if (i < n) p[i] = v;
}
__global__ void k_fill_u32(unsigned* __restrict__ p, unsigned v, int n) {
    int i = blockIdx.x * blockDim.x + threadIdx.x;
    if (i < n) p[i] = v;
}
__global__ void k_cast_h(const float* __restrict__ s, _Float16* __restrict__ d, int n) {
    int i = blockIdx.x * blockDim.x + threadIdx.x;
    if (i < n) d[i] = (_Float16)s[i];
}
// W1 [128,128] f32 -> transposed f16 [c][k]
__global__ void k_w1_t(const float* __restrict__ W1, _Float16* __restrict__ d) {
    int i = blockIdx.x * blockDim.x + threadIdx.x;
    if (i < 128 * 128) {
        int c = i >> 7, k = i & 127;
        d[i] = (_Float16)W1[k * 128 + c];
    }
}
// W2 [128,8] f32 -> transposed padded f16 [16][128] (cols 8..15 zero)
__global__ void k_w2_t(const float* __restrict__ W2, _Float16* __restrict__ d) {
    int i = blockIdx.x * blockDim.x + threadIdx.x;
    if (i < 16 * 128) {
        int c = i >> 7, k = i & 127;
        d[i] = (_Float16)(c < 8 ? W2[k * 8 + c] : 0.f);
    }
}

// ---------- WMMA GEMM: C[M x NC] = A[M x 128] * B[128 x NC] ----------
// A row-major f16; Bt = B transposed f16 [NC][128]. Block = 256 thr (8 waves),
// 128-row M tile, full K resident in LDS, one barrier. f32 accumulate.
template <int NC>
__global__ __launch_bounds__(256) void k_gemm_wmma(const _Float16* __restrict__ A,
                                                   const _Float16* __restrict__ Bt,
                                                   float* __restrict__ C, int M) {
    __shared__ alignas(32) _Float16 sA[128 * 128];   // [row][k]
    __shared__ alignas(32) _Float16 sB[NC * 128];    // [col][k]
    const int t  = threadIdx.x;
    const int m0 = blockIdx.x * 128;
    const bool full = (m0 + 128 <= M);   // uniform: all rows in range?

    // cooperative tile loads, 16B per transaction (async-to-LDS when available)
    {
        uint4* dA = (uint4*)sA;
        const uint4* gA = (const uint4*)A;  // A row = 16 uint4
        for (int i = t; i < 128 * 16; i += 256) {
            int gr = m0 + (i >> 4);
            if (gr < M) copy16_g2l(&dA[i], &gA[(size_t)gr * 16 + (i & 15)]);
            else        dA[i] = uint4{0u, 0u, 0u, 0u};
        }
        uint4* dB = (uint4*)sB;
        const uint4* gB = (const uint4*)Bt;
        for (int i = t; i < NC * 16; i += 256) copy16_g2l(&dB[i], &gB[i]);
        copy_wait();
    }
    __syncthreads();

    const int w    = t >> 5;   // wave id 0..7
    const int l    = t & 31;   // lane
    const int row  = l & 15;   // A row in tile / B+C column in tile
    const int half = l >> 4;   // lane half
    const _Float16* aBase = sA + (w * 16 + row) * 128;

    // A fragments for all 4 K-steps (ISA 7.12.2, 16-bit A 16x32: lane-half owns
    // two contiguous 8-wide K chunks at +0 and +16)
    v16h afrag[4];
#pragma unroll
    for (int kb = 0; kb < 4; ++kb) {
        const v8h lo = *(const v8h*)(aBase + kb * 32 + half * 8);
        const v8h hi = *(const v8h*)(aBase + kb * 32 + 16 + half * 8);
        afrag[kb] = __builtin_shufflevector(lo, hi, 0, 1, 2, 3, 4, 5, 6, 7,
                                            8, 9, 10, 11, 12, 13, 14, 15);
    }

#pragma unroll
    for (int nt = 0; nt < NC / 16; ++nt) {
        v8f acc = {};
        const int col = nt * 16 + row;
        const _Float16* bBase = sB + (size_t)col * 128;
#pragma unroll
        for (int kb = 0; kb < 4; ++kb) {
            // B 32x16 layout: lane-half owns two contiguous 8-wide K chunks at +0 and +16
            const v8h blo = *(const v8h*)(bBase + kb * 32 + half * 8);
            const v8h bhi = *(const v8h*)(bBase + kb * 32 + 16 + half * 8);
            const v16h bfrag = __builtin_shufflevector(blo, bhi, 0, 1, 2, 3, 4, 5, 6, 7,
                                                       8, 9, 10, 11, 12, 13, 14, 15);
            acc = __builtin_amdgcn_wmma_f32_16x16x32_f16(false, afrag[kb], false, bfrag,
                                                         (short)0, acc, false, false);
        }
        // C/D layout: VGPR i -> M = 8*half + i, N = lane&15
        const int orow0 = m0 + w * 16 + half * 8;
        float* cp = C + (size_t)orow0 * NC + col;
        if (full) {
#pragma unroll
            for (int i = 0; i < 8; ++i) cp[(size_t)i * NC] = acc[i];
        } else {
#pragma unroll
            for (int i = 0; i < 8; ++i)
                if (orow0 + i < M) cp[(size_t)i * NC] = acc[i];
        }
    }
}

// ---------- attention dot: out[n,h] = sum_c xw[n*STRIDE + h*C + c] * att[h*C + c] ----------
template <int H, int C, int STRIDE>
__global__ void k_attdot(const float* __restrict__ xw, const float* __restrict__ att,
                         float* __restrict__ out, int N) {
    int i = blockIdx.x * blockDim.x + threadIdx.x;
    if (i >= N * H) return;
    int n = i / H, h = i - n * H;
    const float* p = xw + (size_t)n * STRIDE + h * C;
    const float* a = att + h * C;
    float s = 0.f;
#pragma unroll
    for (int c = 0; c < C; ++c) s += p[c] * a[c];
    out[i] = s;
}

// ---------- edge passes ----------
__device__ __forceinline__ void edge_sd(const int* __restrict__ ei, int eid, int E,
                                        int& s, int& d) {
    if (eid < E) { s = ei[eid]; d = ei[E + eid]; }
    else         { s = d = eid - E; }
}

template <int H>
__global__ void k_edge_max(const int* __restrict__ ei, const float* __restrict__ as,
                           const float* __restrict__ ad, unsigned* __restrict__ emax,
                           int E, int N) {
    int i = blockIdx.x * blockDim.x + threadIdx.x;
    int tot = (E + N) * H;
    if (i >= tot) return;
    int eid = i / H, h = i - eid * H;
    int s, d; edge_sd(ei, eid, E, s, d);
    float v = as[s * H + h] + ad[d * H + h];
    v = v > 0.f ? v : NEG_SLOPE * v;
    atomicMax(&emax[d * H + h], fenc(v));
}

template <int H>
__global__ void k_edge_exp(const int* __restrict__ ei, const float* __restrict__ as,
                           const float* __restrict__ ad, const unsigned* __restrict__ emax,
                           float* __restrict__ den, float* __restrict__ eexp, int E, int N) {
    int i = blockIdx.x * blockDim.x + threadIdx.x;
    int tot = (E + N) * H;
    if (i >= tot) return;
    int eid = i / H, h = i - eid * H;
    int s, d; edge_sd(ei, eid, E, s, d);
    float v = as[s * H + h] + ad[d * H + h];
    v = v > 0.f ? v : NEG_SLOPE * v;
    float ex = expf(v - fdec(emax[d * H + h]));
    eexp[i] = ex;
    atomicAdd(&den[d * H + h], ex);
}

// layer-1 aggregation: one wave per edge; lane pass j handles channel lane+32j (head j)
__global__ void k_agg1(const int* __restrict__ ei, const float* __restrict__ eexp,
                       const float* __restrict__ den, const float* __restrict__ xw,
                       float* __restrict__ agg, int E, int N) {
    int gid = blockIdx.x * blockDim.x + threadIdx.x;
    int eid = gid >> 5, l = gid & 31;
    if (eid >= E + N) return;
    int s, d; edge_sd(ei, eid, E, s, d);
    const float* srow = xw + (size_t)s * 128;
    __builtin_prefetch(srow + l, 0, 0);           // global_prefetch_b8 hint for the gather
    const float4 ex4 = ((const float4*)eexp)[eid];
    const float4 dn4 = ((const float4*)den)[d];
    float alpha[4] = {ex4.x / dn4.x, ex4.y / dn4.y, ex4.z / dn4.z, ex4.w / dn4.w};
#pragma unroll
    for (int j = 0; j < 4; ++j) {
        int c = l + 32 * j;
        atomicAdd(&agg[(size_t)d * 128 + c], alpha[j] * srow[c]);
    }
}

// layer-2 aggregation: one thread per edge, 8 channels (xw2 row stride = 16)
__global__ void k_agg2(const int* __restrict__ ei, const float* __restrict__ eexp,
                       const float* __restrict__ den, const float* __restrict__ xw2,
                       float* __restrict__ agg, int E, int N) {
    int eid = blockIdx.x * blockDim.x + threadIdx.x;
    if (eid >= E + N) return;
    int s, d; edge_sd(ei, eid, E, s, d);
    float alpha = eexp[eid] / den[d];
#pragma unroll
    for (int c = 0; c < 8; ++c)
        atomicAdd(&agg[(size_t)d * 8 + c], alpha * xw2[(size_t)s * 16 + c]);
}

// h = relu(agg1 + b1) -> f16 for GEMM2
__global__ void k_bias_relu(const float* __restrict__ agg, const float* __restrict__ b,
                            _Float16* __restrict__ h16, int N) {
    int i = blockIdx.x * blockDim.x + threadIdx.x;
    if (i >= N * 128) return;
    float v = agg[i] + b[i & 127];
    h16[i] = (_Float16)(v > 0.f ? v : 0.f);
}

// out = log_softmax(agg2 + b2) over 8 classes
__global__ void k_final(const float* __restrict__ agg2, const float* __restrict__ b2,
                        float* __restrict__ out, int N) {
    int n = blockIdx.x * blockDim.x + threadIdx.x;
    if (n >= N) return;
    float o[8], m = -3.4e38f;
#pragma unroll
    for (int c = 0; c < 8; ++c) { o[c] = agg2[n * 8 + c] + b2[c]; m = fmaxf(m, o[c]); }
    float s = 0.f;
#pragma unroll
    for (int c = 0; c < 8; ++c) s += expf(o[c] - m);
    float lg = logf(s);
#pragma unroll
    for (int c = 0; c < 8; ++c) out[n * 8 + c] = o[c] - m - lg;
}

// ---------- host ----------
static inline unsigned cdivu(long long a, long long b) { return (unsigned)((a + b - 1) / b); }

extern "C" void kernel_launch(void* const* d_in, const int* in_sizes, int n_in,
                              void* d_out, int out_size, void* d_ws, size_t ws_size,
                              hipStream_t stream) {
    const float* x        = (const float*)d_in[0];
    const int*   ei       = (const int*)d_in[1];
    const float* W1       = (const float*)d_in[2];
    const float* att_src1 = (const float*)d_in[3];
    const float* att_dst1 = (const float*)d_in[4];
    const float* b1       = (const float*)d_in[5];
    const float* W2       = (const float*)d_in[6];
    const float* att_src2 = (const float*)d_in[7];
    const float* att_dst2 = (const float*)d_in[8];
    const float* b2       = (const float*)d_in[9];
    float* out = (float*)d_out;

    const int N = in_sizes[0] / 128;
    const int E = in_sizes[1] / 2;
    const int ET = E + N;  // with self loops

    size_t off = 0;
    auto carve = [&](size_t bytes) -> void* {
        void* p = (char*)d_ws + off;
        off += (bytes + 255) & ~(size_t)255;
        return p;
    };
    _Float16* xh    = (_Float16*)carve((size_t)N * 128 * 2);
    _Float16* w1t   = (_Float16*)carve(128 * 128 * 2);
    _Float16* w2t   = (_Float16*)carve(16 * 128 * 2);
    float*    xw1   = (float*)carve((size_t)N * 128 * 4);
    float*    asrc1 = (float*)carve((size_t)N * 4 * 4);
    float*    adst1 = (float*)carve((size_t)N * 4 * 4);
    unsigned* emax1 = (unsigned*)carve((size_t)N * 4 * 4);
    float*    den1  = (float*)carve((size_t)N * 4 * 4);
    float*    eexp1 = (float*)carve((size_t)ET * 4 * 4);
    float*    agg1  = (float*)carve((size_t)N * 128 * 4);
    _Float16* h16   = (_Float16*)carve((size_t)N * 128 * 2);
    float*    xw2   = (float*)carve((size_t)N * 16 * 4);
    float*    asrc2 = (float*)carve((size_t)N * 4);
    float*    adst2 = (float*)carve((size_t)N * 4);
    unsigned* emax2 = (unsigned*)carve((size_t)N * 4);
    float*    den2  = (float*)carve((size_t)N * 4);
    float*    eexp2 = (float*)carve((size_t)ET * 4);
    float*    agg2  = (float*)carve((size_t)N * 8 * 4);
    (void)ws_size; (void)n_in; (void)out_size;

    const unsigned B = 256;

    // init accumulators
    k_fill_u32<<<cdivu((long long)N * 4, B), B, 0, stream>>>(emax1, ENC_NEG_INF, N * 4);
    k_fill_f32<<<cdivu((long long)N * 4, B), B, 0, stream>>>(den1, 0.f, N * 4);
    k_fill_f32<<<cdivu((long long)N * 128, B), B, 0, stream>>>(agg1, 0.f, N * 128);
    k_fill_u32<<<cdivu(N, B), B, 0, stream>>>(emax2, ENC_NEG_INF, N);
    k_fill_f32<<<cdivu(N, B), B, 0, stream>>>(den2, 0.f, N);
    k_fill_f32<<<cdivu((long long)N * 8, B), B, 0, stream>>>(agg2, 0.f, N * 8);

    // f16 operand prep (weights stored transposed for contiguous WMMA B-fragments)
    k_cast_h<<<cdivu((long long)N * 128, B), B, 0, stream>>>(x, xh, N * 128);
    k_w1_t<<<cdivu(128 * 128, B), B, 0, stream>>>(W1, w1t);
    k_w2_t<<<cdivu(16 * 128, B), B, 0, stream>>>(W2, w2t);

    // layer 1: xw1 = x @ W1 (WMMA), attention scalars
    k_gemm_wmma<128><<<cdivu(N, 128), B, 0, stream>>>(xh, w1t, xw1, N);
    k_attdot<4, 32, 128><<<cdivu((long long)N * 4, B), B, 0, stream>>>(xw1, att_src1, asrc1, N);
    k_attdot<4, 32, 128><<<cdivu((long long)N * 4, B), B, 0, stream>>>(xw1, att_dst1, adst1, N);

    // layer 1: segment softmax + aggregate
    k_edge_max<4><<<cdivu((long long)ET * 4, B), B, 0, stream>>>(ei, asrc1, adst1, emax1, E, N);
    k_edge_exp<4><<<cdivu((long long)ET * 4, B), B, 0, stream>>>(ei, asrc1, adst1, emax1, den1, eexp1, E, N);
    k_agg1<<<cdivu((long long)ET * 32, B), B, 0, stream>>>(ei, eexp1, den1, xw1, agg1, E, N);

    // relu(+bias) -> h16
    k_bias_relu<<<cdivu((long long)N * 128, B), B, 0, stream>>>(agg1, b1, h16, N);

    // layer 2: xw2 = h @ W2 (WMMA, padded to 16 cols), attention scalars
    k_gemm_wmma<16><<<cdivu(N, 128), B, 0, stream>>>(h16, w2t, xw2, N);
    k_attdot<1, 8, 16><<<cdivu(N, B), B, 0, stream>>>(xw2, att_src2, asrc2, N);
    k_attdot<1, 8, 16><<<cdivu(N, B), B, 0, stream>>>(xw2, att_dst2, adst2, N);

    // layer 2: segment softmax + aggregate
    k_edge_max<1><<<cdivu(ET, B), B, 0, stream>>>(ei, asrc2, adst2, emax2, E, N);
    k_edge_exp<1><<<cdivu(ET, B), B, 0, stream>>>(ei, asrc2, adst2, emax2, den2, eexp2, E, N);
    k_agg2<<<cdivu(ET, B), B, 0, stream>>>(ei, eexp2, den2, xw2, agg2, E, N);

    // bias + log_softmax
    k_final<<<cdivu(N, B), B, 0, stream>>>(agg2, b2, out, N);
}
